// TransformerEncoderLayer_27977416966159
// MI455X (gfx1250) — compile-verified, matrix operand order checked by reading, not run on hardware
//
#include <hip/hip_runtime.h>
#include <hip/hip_bf16.h>

// ---------------------------------------------------------------------------
// Transformer encoder layer for MI455X (gfx1250), wave32 + WMMA bf16 +
// async global->LDS staging (double-buffered) when the toolchain exposes it.
// B=2, S=2048, D_MODEL=1024, H=16, Dk=64, D_FF=4096. All dims tile-divisible.
// ---------------------------------------------------------------------------

#define BATCH   2
#define SEQ     2048
#define DMODEL  1024
#define NH      16
#define DK      64
#define DFF     4096
#define MROWS   (BATCH * SEQ)   // 4096

typedef __attribute__((ext_vector_type(16))) __bf16 v16bf;
typedef __attribute__((ext_vector_type(8)))  float  v8f;
typedef __attribute__((ext_vector_type(4)))  int    v4i;

union FragBF { v16bf v; unsigned u[8]; };
union BF4    { __bf16 h[4]; uint2 u2; };

#define WMMA_BF16(A, B, C) \
  __builtin_amdgcn_wmma_f32_16x16x32_bf16(false, (A), false, (B), (short)0, (C), false, false)

// 16-bit A/B matrix VGPR layout (ISA 7.12.2): VGPR pair j holds K =
// (j<4 ? 0 : 16) + (lane>>4)*8 + (j&3)*2 .. +1  (contiguous bf16 pair)
__device__ __forceinline__ int kpair(int j, int kg) {
  return ((j & 4) << 2) + (kg << 3) + ((j & 3) << 1);
}

// --- CDNA5 async global->LDS path (GLOBAL_LOAD_ASYNC_TO_LDS_B128, ASYNCcnt) ---
#if defined(__AMDGCN__) && \
    __has_builtin(__builtin_amdgcn_global_load_async_to_lds_b128) && \
    __has_builtin(__builtin_amdgcn_s_wait_asynccnt)
#define USE_ASYNC_LDS 1
#else
#define USE_ASYNC_LDS 0
#endif

#if USE_ASYNC_LDS
typedef __attribute__((address_space(1))) v4i* g4ptr;   // global int4*
typedef __attribute__((address_space(3))) v4i* l4ptr;   // LDS int4*
#endif

// Copy 16 contiguous bf16 (32B) from global to LDS.
__device__ __forceinline__ void stage16(__bf16* dst_lds, const __bf16* src_g) {
#if USE_ASYNC_LDS
  g4ptr g0 = (g4ptr)(uintptr_t)(src_g);
  g4ptr g1 = (g4ptr)(uintptr_t)(src_g + 8);
  l4ptr l0 = (l4ptr)(unsigned)(uintptr_t)(dst_lds);
  l4ptr l1 = (l4ptr)(unsigned)(uintptr_t)(dst_lds + 8);
  __builtin_amdgcn_global_load_async_to_lds_b128(g0, l0, 0, 0);
  __builtin_amdgcn_global_load_async_to_lds_b128(g1, l1, 0, 0);
#else
  float4 a0 = *(const float4*)(src_g);
  float4 a1 = *(const float4*)(src_g + 8);
  *(float4*)(dst_lds)     = a0;
  *(float4*)(dst_lds + 8) = a1;
#endif
}

__device__ __forceinline__ void async_stage_wait() {
#if USE_ASYNC_LDS
  __builtin_amdgcn_s_wait_asynccnt(0);
#endif
}

// ---------------------------------------------------------------------------
// fp32 -> bf16 elementwise convert (4 elems/thread)
// ---------------------------------------------------------------------------
__global__ void cvt_f32_bf16_kernel(const float* __restrict__ in,
                                    __bf16* __restrict__ out, int n) {
  int i = (blockIdx.x * blockDim.x + threadIdx.x) * 4;
  if (i + 3 < n) {
    float4 f = *(const float4*)(in + i);
    out[i + 0] = (__bf16)f.x;
    out[i + 1] = (__bf16)f.y;
    out[i + 2] = (__bf16)f.z;
    out[i + 3] = (__bf16)f.w;
  }
}

// ---------------------------------------------------------------------------
// Transposing convert: out[N][K] bf16 = in[K][N] fp32. 32x32 LDS tile.
// grid = (K/32, N/32), 256 threads.
// ---------------------------------------------------------------------------
__global__ __launch_bounds__(256) void cvt_transpose_kernel(
    const float* __restrict__ in, __bf16* __restrict__ out, int K, int N) {
  __shared__ __bf16 t[32][33];
  const int k0 = blockIdx.x * 32, n0 = blockIdx.y * 32;
  const int tid = threadIdx.x;
  const int r  = tid >> 3;         // 0..31
  const int c4 = (tid & 7) * 4;    // 0,4,...,28
  float4 f = *(const float4*)(in + (size_t)(k0 + r) * N + n0 + c4);
  t[r][c4 + 0] = (__bf16)f.x;
  t[r][c4 + 1] = (__bf16)f.y;
  t[r][c4 + 2] = (__bf16)f.z;
  t[r][c4 + 3] = (__bf16)f.w;
  __syncthreads();
  BF4 w;
  w.h[0] = t[c4 + 0][r];
  w.h[1] = t[c4 + 1][r];
  w.h[2] = t[c4 + 2][r];
  w.h[3] = t[c4 + 3][r];
  *(uint2*)(out + (size_t)(n0 + r) * K + k0 + c4) = w.u2;
}

// ---------------------------------------------------------------------------
// bf16 GEMM: C[M,N] = A[M,K] @ B[K,N] with B passed TRANSPOSED (Bt[N][K]).
// Block = 256 threads (8 waves), tile 128x128, K-step 32, double-buffered LDS
// with async global->LDS staging. Wave grid 2(M) x 4(N): per-wave 64x32.
// ---------------------------------------------------------------------------
constexpr int FLAG_BIAS   = 1;
constexpr int FLAG_GELU   = 2;
constexpr int FLAG_BF16   = 4;
constexpr int FLAG_F32    = 8;
constexpr int FLAG_BF16VT = 16;   // write bf16 output transposed as [B][H][Dk][S]

template <int FLAGS>
__global__ __launch_bounds__(256) void gemm_bf16_kernel(
    const __bf16* __restrict__ A, const __bf16* __restrict__ Bt,
    const float* __restrict__ bias, float* __restrict__ Cf,
    __bf16* __restrict__ Cb, int M, int N, int K) {
  __shared__ __bf16 As[2][128 * 32];   // [row][k]
  __shared__ __bf16 Bs[2][128 * 32];   // [col][k]

  const int tid  = threadIdx.x;
  const int lane = tid & 31;
  const int wave = tid >> 5;
  const int l15  = lane & 15;
  const int kg   = lane >> 4;
  const int wm   = wave & 1;        // M direction (64 rows each)
  const int wn   = wave >> 1;       // N direction (32 cols each)
  const int mblk = blockIdx.y * 128;
  const int nblk = blockIdx.x * 128;

  v8f acc[4][2];
#pragma unroll
  for (int ms = 0; ms < 4; ++ms)
#pragma unroll
    for (int ns = 0; ns < 2; ++ns) acc[ms][ns] = {};

  // staging: thread t covers (row = t>>1, 16 elems at (t&1)*16) of a 128x32 tile
  const int srow = tid >> 1, scol = (tid & 1) * 16;
  const __bf16* ag = A  + (size_t)(mblk + srow) * K + scol;
  const __bf16* bg = Bt + (size_t)(nblk + srow) * K + scol;

  stage16(&As[0][srow * 32 + scol], ag);
  stage16(&Bs[0][srow * 32 + scol], bg);
  async_stage_wait();
  __syncthreads();

  const int nk = K >> 5;
  for (int kt = 0; kt < nk; ++kt) {
    const int cur = kt & 1, nxt = cur ^ 1;
    if (kt + 1 < nk) {
      stage16(&As[nxt][srow * 32 + scol], ag + (kt + 1) * 32);
      stage16(&Bs[nxt][srow * 32 + scol], bg + (kt + 1) * 32);
    }

    FragBF af[4], bf[2];
#pragma unroll
    for (int ms = 0; ms < 4; ++ms)
#pragma unroll
      for (int j = 0; j < 8; ++j)
        af[ms].u[j] =
            *(const unsigned*)(&As[cur][(wm * 64 + ms * 16 + l15) * 32 + kpair(j, kg)]);
#pragma unroll
    for (int ns = 0; ns < 2; ++ns)
#pragma unroll
      for (int j = 0; j < 8; ++j)
        bf[ns].u[j] =
            *(const unsigned*)(&Bs[cur][(wn * 32 + ns * 16 + l15) * 32 + kpair(j, kg)]);
#pragma unroll
    for (int ms = 0; ms < 4; ++ms)
#pragma unroll
      for (int ns = 0; ns < 2; ++ns)
        acc[ms][ns] = WMMA_BF16(af[ms].v, bf[ns].v, acc[ms][ns]);

    async_stage_wait();
    __syncthreads();
  }

  // Epilogue. C layout: vgpr r -> row = (lane>>4)*8 + r, col = lane&15
#pragma unroll
  for (int ms = 0; ms < 4; ++ms)
#pragma unroll
    for (int ns = 0; ns < 2; ++ns) {
      const int col = nblk + wn * 32 + ns * 16 + l15;
      float bv = 0.f;
      if (FLAGS & FLAG_BIAS) bv = bias[col];
#pragma unroll
      for (int r = 0; r < 8; ++r) {
        const int row = mblk + wm * 64 + ms * 16 + kg * 8 + r;
        float v = acc[ms][ns][r] + bv;
        if (FLAGS & FLAG_GELU) v = 0.5f * v * (1.f + erff(v * 0.70710678118f));
        if (FLAGS & FLAG_F32)  Cf[(size_t)row * N + col] = v;
        if (FLAGS & FLAG_BF16) Cb[(size_t)row * N + col] = (__bf16)v;
        if (FLAGS & FLAG_BF16VT) {
          const int bb = row / SEQ, ss = row - bb * SEQ;
          const int hh = col >> 6, dd = col & (DK - 1);
          Cb[(((size_t)bb * NH + hh) * DK + dd) * SEQ + ss] = (__bf16)v;
        }
      }
    }
}

// ---------------------------------------------------------------------------
// Flash attention: grid (S/64, H, B), 128 threads (4 waves).
// Q/K bf16 [B*S, DMODEL] (head h = cols h*64..); V bf16 pre-transposed
// [B][H][Dk][S]. Key blocks of 32 double-buffered in LDS via async staging.
// ---------------------------------------------------------------------------
__global__ __launch_bounds__(128) void attention_kernel(
    const __bf16* __restrict__ Q, const __bf16* __restrict__ Kmat,
    const __bf16* __restrict__ Vt_g, const int* __restrict__ mask,
    float* __restrict__ ctx) {
  __shared__ __bf16 Ks[2][32 * 64];    // key block, [krow][dk]
  __shared__ __bf16 Vs[2][64 * 32];    // value block transposed, [dk][krow]
  __shared__ __bf16 Ps[4][16 * 32];    // per-wave P tile (C-layout -> A-layout)

  const int tid  = threadIdx.x;
  const int lane = tid & 31;
  const int wave = tid >> 5;
  const int l15  = lane & 15;
  const int kg   = lane >> 4;
  const int h    = blockIdx.y;
  const int b    = blockIdx.z;
  const int qbase = blockIdx.x * 64 + wave * 16;

  // Preload Q fragments (16x64 -> two K=32 chunks), A-matrix layout.
  const __bf16* qrow = Q + ((size_t)(b * SEQ + qbase + l15)) * DMODEL + h * DK;
  FragBF qa[2];
#pragma unroll
  for (int kc = 0; kc < 2; ++kc)
#pragma unroll
    for (int j = 0; j < 8; ++j)
      qa[kc].u[j] = *(const unsigned*)(qrow + kc * 32 + kpair(j, kg));

  float mrun[8], lrun[8];
  v8f acc[4];
#pragma unroll
  for (int r = 0; r < 8; ++r) { mrun[r] = -1e30f; lrun[r] = 0.f; }
#pragma unroll
  for (int nb = 0; nb < 4; ++nb) acc[nb] = {};

  // K stage: thread -> (krow = tid>>2, 16 elems at (tid&3)*16)   [32x64]
  const int krow = tid >> 2, kcol = (tid & 3) * 16;
  // V stage: thread -> (dk = tid>>1, 16 elems at (tid&1)*16)     [64x32]
  const int vrow = tid >> 1, vcol = (tid & 1) * 16;
  const __bf16* kbase = Kmat + ((size_t)(b * SEQ + krow)) * DMODEL + h * DK + kcol;
  const __bf16* vbase = Vt_g + (((size_t)b * NH + h) * DK + vrow) * SEQ + vcol;

  stage16(&Ks[0][krow * 64 + kcol], kbase);
  stage16(&Vs[0][vrow * 32 + vcol], vbase);
  async_stage_wait();
  __syncthreads();

  const int njb = SEQ / 32;
  for (int jb = 0; jb < njb; ++jb) {
    const int j0 = jb * 32;
    const int cur = jb & 1, nxt = cur ^ 1;
    if (jb + 1 < njb) {
      stage16(&Ks[nxt][krow * 64 + kcol], kbase + (size_t)(j0 + 32) * DMODEL);
      stage16(&Vs[nxt][vrow * 32 + vcol], vbase + (j0 + 32));
    }

    // scores: two 16x16 tiles (keys j0..j0+15, j0+16..j0+31)
    v8f s[2];
#pragma unroll
    for (int t = 0; t < 2; ++t) {
      v8f z = {};
      FragBF bk;
#pragma unroll
      for (int j = 0; j < 8; ++j)
        bk.u[j] = *(const unsigned*)(&Ks[cur][(t * 16 + l15) * 64 + kpair(j, kg)]);
      z = WMMA_BF16(qa[0].v, bk.v, z);
#pragma unroll
      for (int j = 0; j < 8; ++j)
        bk.u[j] = *(const unsigned*)(&Ks[cur][(t * 16 + l15) * 64 + 32 + kpair(j, kg)]);
      z = WMMA_BF16(qa[1].v, bk.v, z);
      const int mv = mask[b * SEQ + j0 + t * 16 + l15];
#pragma unroll
      for (int r = 0; r < 8; ++r) {
        float sv = z[r] * 0.125f;            // 1/sqrt(64)
        z[r] = mv ? sv : -1e9f;
      }
      s[t] = z;
    }

    // online softmax (rows live in 16-lane halves -> xor 1,2,4,8)
    float p0[8], p1[8];
#pragma unroll
    for (int r = 0; r < 8; ++r) {
      float tm = fmaxf(s[0][r], s[1][r]);
      tm = fmaxf(tm, __shfl_xor(tm, 1, 32));
      tm = fmaxf(tm, __shfl_xor(tm, 2, 32));
      tm = fmaxf(tm, __shfl_xor(tm, 4, 32));
      tm = fmaxf(tm, __shfl_xor(tm, 8, 32));
      const float mn    = fmaxf(mrun[r], tm);
      const float alpha = __expf(mrun[r] - mn);
      p0[r] = __expf(s[0][r] - mn);
      p1[r] = __expf(s[1][r] - mn);
      float rs = p0[r] + p1[r];
      rs += __shfl_xor(rs, 1, 32);
      rs += __shfl_xor(rs, 2, 32);
      rs += __shfl_xor(rs, 4, 32);
      rs += __shfl_xor(rs, 8, 32);
      lrun[r] = lrun[r] * alpha + rs;
      mrun[r] = mn;
#pragma unroll
      for (int nb = 0; nb < 4; ++nb) acc[nb][r] *= alpha;
    }

    // P (16x32, C-layout) -> per-wave LDS -> A-layout fragment
    __bf16* P = &Ps[wave][0];
#pragma unroll
    for (int r = 0; r < 8; ++r) {
      P[(kg * 8 + r) * 32 + l15]      = (__bf16)p0[r];
      P[(kg * 8 + r) * 32 + 16 + l15] = (__bf16)p1[r];
    }
    FragBF pa;
#pragma unroll
    for (int j = 0; j < 8; ++j)
      pa.u[j] = *(const unsigned*)(&P[l15 * 32 + kpair(j, kg)]);

    // ctx += P (16x32) @ V (32x64)
#pragma unroll
    for (int nb = 0; nb < 4; ++nb) {
      FragBF bv;
#pragma unroll
      for (int j = 0; j < 8; ++j)
        bv.u[j] = *(const unsigned*)(&Vs[cur][(nb * 16 + l15) * 32 + kpair(j, kg)]);
      acc[nb] = WMMA_BF16(pa.v, bv.v, acc[nb]);
    }

    async_stage_wait();
    __syncthreads();
  }

  // normalize and write ctx as [B*S, DMODEL] (head h slice)
#pragma unroll
  for (int r = 0; r < 8; ++r) {
    const float inv = 1.f / lrun[r];
    const size_t row = (size_t)(b * SEQ + qbase + kg * 8 + r);
#pragma unroll
    for (int nb = 0; nb < 4; ++nb)
      ctx[row * DMODEL + h * DK + nb * 16 + l15] = acc[nb][r] * inv;
  }
}

// ---------------------------------------------------------------------------
// out = LayerNorm(X + Y) * g + be ; one block (256 thr) per row of 1024.
// Optionally also emits a bf16 copy for the following GEMM.
// ---------------------------------------------------------------------------
__global__ __launch_bounds__(256) void add_ln_kernel(
    const float* __restrict__ X, const float* __restrict__ Y,
    const float* __restrict__ g, const float* __restrict__ be,
    float* __restrict__ outf, __bf16* __restrict__ outb) {
  __shared__ float red[16];
  const int row = blockIdx.x, tid = threadIdx.x;
  const int lane = tid & 31, wave = tid >> 5;
  const float* xr = X + (size_t)row * DMODEL;
  const float* yr = Y + (size_t)row * DMODEL;
  float v[4], s = 0.f, s2 = 0.f;
#pragma unroll
  for (int i = 0; i < 4; ++i) {
    const int c = tid + i * 256;
    v[i] = xr[c] + yr[c];
    s  += v[i];
    s2 += v[i] * v[i];
  }
#pragma unroll
  for (int m = 1; m < 32; m <<= 1) {
    s  += __shfl_xor(s, m, 32);
    s2 += __shfl_xor(s2, m, 32);
  }
  if (lane == 0) { red[wave] = s; red[8 + wave] = s2; }
  __syncthreads();
  if (wave == 0) {
    float a  = (lane < 8) ? red[lane] : 0.f;
    float b2 = (lane < 8) ? red[8 + lane] : 0.f;
#pragma unroll
    for (int m = 1; m < 8; m <<= 1) {
      a  += __shfl_xor(a, m, 32);
      b2 += __shfl_xor(b2, m, 32);
    }
    if (lane == 0) { red[0] = a; red[8] = b2; }
  }
  __syncthreads();
  const float mean = red[0] * (1.f / DMODEL);
  const float var  = red[8] * (1.f / DMODEL) - mean * mean;
  const float rstd = rsqrtf(var + 1e-6f);
#pragma unroll
  for (int i = 0; i < 4; ++i) {
    const int c = tid + i * 256;
    const float o = (v[i] - mean) * rstd * g[c] + be[c];
    outf[(size_t)row * DMODEL + c] = o;
    if (outb) outb[(size_t)row * DMODEL + c] = (__bf16)o;
  }
}

// ---------------------------------------------------------------------------
// Host-side pipeline
// ---------------------------------------------------------------------------
extern "C" void kernel_launch(void* const* d_in, const int* in_sizes, int n_in,
                              void* d_out, int out_size, void* d_ws, size_t ws_size,
                              hipStream_t stream) {
  const float* x    = (const float*)d_in[0];
  const int*   mask = (const int*)d_in[1];
  const float* wq   = (const float*)d_in[2];
  const float* wk   = (const float*)d_in[3];
  const float* wv   = (const float*)d_in[4];
  const float* wo   = (const float*)d_in[5];
  const float* wo_b = (const float*)d_in[6];
  const float* w1   = (const float*)d_in[7];
  const float* b1   = (const float*)d_in[8];
  const float* w2   = (const float*)d_in[9];
  const float* b2   = (const float*)d_in[10];
  const float* g1   = (const float*)d_in[11];
  const float* be1  = (const float*)d_in[12];
  const float* g2   = (const float*)d_in[13];
  const float* be2  = (const float*)d_in[14];

  char* ws = (char*)d_ws;
  size_t off = 0;
  auto alloc = [&](size_t bytes) -> void* {
    void* p = ws + off;
    off = (off + bytes + 255) & ~(size_t)255;
    return p;
  };

  const size_t MD = (size_t)MROWS * DMODEL;   // 4M
  const size_t DD = (size_t)DMODEL * DMODEL;  // 1M
  const size_t DF = (size_t)DMODEL * DFF;     // 4M
  const size_t MF = (size_t)MROWS * DFF;      // 16M

  __bf16* xb   = (__bf16*)alloc(MD * 2);
  __bf16* wqT  = (__bf16*)alloc(DD * 2);      // [N][K] transposed weights
  __bf16* wkT  = (__bf16*)alloc(DD * 2);
  __bf16* wvT  = (__bf16*)alloc(DD * 2);
  __bf16* woT  = (__bf16*)alloc(DD * 2);
  __bf16* w1T  = (__bf16*)alloc(DF * 2);
  __bf16* w2T  = (__bf16*)alloc(DF * 2);
  __bf16* Qb   = (__bf16*)alloc(MD * 2);
  __bf16* Kb   = (__bf16*)alloc(MD * 2);
  __bf16* VbT  = (__bf16*)alloc(MD * 2);      // [B][H][Dk][S]
  float*  ctx  = (float*)alloc(MD * 4);
  __bf16* ctxb = (__bf16*)alloc(MD * 2);
  float*  attn = (float*)alloc(MD * 4);
  float*  h1   = (float*)alloc(MD * 4);
  __bf16* h1b  = (__bf16*)alloc(MD * 2);
  __bf16* ffb  = (__bf16*)alloc(MF * 2);
  float*  ff2  = (float*)alloc(MD * 4);

  // --- precision demotion: x plain, weights transposed to [N][K] bf16 ---
  cvt_f32_bf16_kernel<<<MD / 1024, 256, 0, stream>>>(x, xb, (int)MD);
  dim3 tDD(DMODEL / 32, DMODEL / 32);
  cvt_transpose_kernel<<<tDD, 256, 0, stream>>>(wq, wqT, DMODEL, DMODEL);
  cvt_transpose_kernel<<<tDD, 256, 0, stream>>>(wk, wkT, DMODEL, DMODEL);
  cvt_transpose_kernel<<<tDD, 256, 0, stream>>>(wv, wvT, DMODEL, DMODEL);
  cvt_transpose_kernel<<<tDD, 256, 0, stream>>>(wo, woT, DMODEL, DMODEL);
  cvt_transpose_kernel<<<dim3(DMODEL / 32, DFF / 32), 256, 0, stream>>>(
      w1, w1T, DMODEL, DFF);
  cvt_transpose_kernel<<<dim3(DFF / 32, DMODEL / 32), 256, 0, stream>>>(
      w2, w2T, DFF, DMODEL);

  // --- QKV projections (bf16 out; V written pre-transposed for attention) ---
  dim3 gqkv(DMODEL / 128, MROWS / 128);
  gemm_bf16_kernel<FLAG_BF16><<<gqkv, 256, 0, stream>>>(
      xb, wqT, nullptr, nullptr, Qb, MROWS, DMODEL, DMODEL);
  gemm_bf16_kernel<FLAG_BF16><<<gqkv, 256, 0, stream>>>(
      xb, wkT, nullptr, nullptr, Kb, MROWS, DMODEL, DMODEL);
  gemm_bf16_kernel<FLAG_BF16VT><<<gqkv, 256, 0, stream>>>(
      xb, wvT, nullptr, nullptr, VbT, MROWS, DMODEL, DMODEL);

  // --- flash attention ---
  attention_kernel<<<dim3(SEQ / 64, NH, BATCH), 128, 0, stream>>>(
      Qb, Kb, VbT, mask, ctx);
  cvt_f32_bf16_kernel<<<MD / 1024, 256, 0, stream>>>(ctx, ctxb, (int)MD);

  // --- output projection + bias ---
  gemm_bf16_kernel<FLAG_BIAS | FLAG_F32><<<gqkv, 256, 0, stream>>>(
      ctxb, woT, wo_b, attn, nullptr, MROWS, DMODEL, DMODEL);

  // --- residual + LN1 (fp32 + bf16 views) ---
  add_ln_kernel<<<MROWS, 256, 0, stream>>>(x, attn, g1, be1, h1, h1b);

  // --- FFN: gelu(h1 @ w1 + b1) @ w2 + b2 ---
  gemm_bf16_kernel<FLAG_BIAS | FLAG_GELU | FLAG_BF16>
      <<<dim3(DFF / 128, MROWS / 128), 256, 0, stream>>>(
          h1b, w1T, b1, nullptr, ffb, MROWS, DFF, DMODEL);
  gemm_bf16_kernel<FLAG_BIAS | FLAG_F32><<<gqkv, 256, 0, stream>>>(
      ffb, w2T, b2, ff2, nullptr, MROWS, DMODEL, DFF);

  // --- residual + LN2 -> output ---
  add_ln_kernel<<<MROWS, 256, 0, stream>>>(h1, ff2, g2, be2, (float*)d_out, nullptr);
}
